// CrossViewImputer_16587163697656
// MI455X (gfx1250) — matrix operand order repaired and checked by reading, compiled, not run on hardware
//
#include <hip/hip_runtime.h>
#include <hip/hip_bf16.h>

// ---------------------------------------------------------------------------
// GAT (2 layers) + MLP decoder for MI455X (gfx1250, wave32, WMMA).
// GEMMs: v_wmma_f32_16x16x32_bf16, fully unrolled K, 4 N-tiles per wave,
// b128 fragment loads (B pre-packed into fragment order), B fragments for a
// kstep loaded as one clause ahead of the 4 WMMAs (load/compute overlap).
// Graph aggregation: coalesced gather + f32 L2 atomics (bandwidth-bound part).
// ---------------------------------------------------------------------------

typedef __attribute__((ext_vector_type(16))) __bf16 v16bf;
typedef __attribute__((ext_vector_type(8)))  float  v8f;

#define H1 4
#define C_DIM 64
#define HC1 256      // H1 * C_DIM
#define IN_DIM 128
#define HID 64
#define NCLS 17

// ---------------- helpers ----------------

__device__ __forceinline__ unsigned short f32_to_bf16_rne(float f) {
    unsigned u = __float_as_uint(f);
    unsigned r = u + 0x7FFFu + ((u >> 16) & 1u);
    return (unsigned short)(r >> 16);
}

// monotonic float <-> uint key (atomicMax-based segment max with negatives)
__device__ __forceinline__ unsigned fkey(float f) {
    unsigned u = __float_as_uint(f);
    return (u & 0x80000000u) ? ~u : (u | 0x80000000u);
}
__device__ __forceinline__ float funkey(unsigned k) {
    unsigned u = (k & 0x80000000u) ? (k ^ 0x80000000u) : ~k;
    return __uint_as_float(u);
}

// ---------------- conversion ----------------

__global__ void k_f32_to_bf16(const float* __restrict__ in,
                              unsigned short* __restrict__ out, int n) {
    int i = blockIdx.x * blockDim.x + threadIdx.x;
    if (i < n) out[i] = f32_to_bf16_rne(in[i]);
}

// ---------------- B-matrix fragment pre-pack --------------------------------
// Packs B[K,N] (bf16 row-major) into WMMA B-fragment order:
//   Bp[((ks*(N/16) + tn)*32 + lane)*8 + r]  (dwords, 2 bf16 each)
// B fragment (32x16, ISA 7.12.2): VGPR r holds K = ks*32 + 8h + 2r,2r+1
// (+8 more for r >= 4; h = lane/16), N = tn*16 + lane%16.

__global__ void k_pack_b(const unsigned short* __restrict__ B,
                         unsigned int* __restrict__ Bp, int K, int N) {
    int idx = blockIdx.x * blockDim.x + threadIdx.x;   // (ks, tn, lane)
    int tilesN = N >> 4;
    int total  = (K >> 5) * tilesN * 32;
    if (idx >= total) return;
    int lane = idx & 31;
    int t    = idx >> 5;
    int tn   = t % tilesN;
    int ks   = t / tilesN;
    int nn   = tn * 16 + (lane & 15);
    int k0   = ks * 32 + ((lane >= 16) ? 8 : 0);
#pragma unroll
    for (int r = 0; r < 8; ++r) {
        unsigned lo = B[(size_t)(k0 + 2 * r + (r >= 4 ? 8 : 0))     * N + nn];
        unsigned hi = B[(size_t)(k0 + 2 * r + (r >= 4 ? 8 : 0) + 1) * N + nn];
        Bp[(size_t)idx * 8 + r] = lo | (hi << 16);
    }
}

// ---------------- WMMA GEMM: C[M,N] = A[M,K] x Bp, bf16 in, f32 out ----------
// One wave computes a 16 x (16*NT) strip. K, N compile-time; K % 32 == 0,
// M % 16 == 0 (M = 50000 = 3125 * 16).
// A fragment: u[0..3] = K  kb+8h .. kb+8h+7   (16B contiguous)
//             u[4..7] = K  kb+16+8h .. +7     (16B contiguous)

template <int K, int N, int NT>
__global__ void k_wmma_gemm(const unsigned short* __restrict__ A,
                            const unsigned int* __restrict__ Bp,
                            float* __restrict__ C, int totalWaves) {
    int w = blockIdx.x * (blockDim.x >> 5) + (threadIdx.x >> 5);
    if (w >= totalWaves) return;                 // wave-uniform: EXEC stays full
    constexpr int strips = N / (16 * NT);
    int tm = w / strips;
    int ts = w - tm * strips;
    int lane = threadIdx.x & 31;
    int half = lane >> 4;
    int l15  = lane & 15;
    int m    = tm * 16 + l15;

    union Frag { uint4 q[2]; v16bf v; };

    v8f acc[NT];
#pragma unroll
    for (int t = 0; t < NT; ++t) acc[t] = (v8f)(0.0f);

    const unsigned short* arow = A + (size_t)m * K + 8 * half;
    // per-lane base of the packed-B stream for this strip
    const uint4* bbase = reinterpret_cast<const uint4*>(
        Bp + ((size_t)(ts * NT) * 32 + lane) * 8);

#pragma unroll
    for (int ks = 0; ks < K / 32; ++ks) {
        Frag a;
        const uint4* ap = reinterpret_cast<const uint4*>(arow + ks * 32);
        a.q[0] = ap[0];                          // K = kb+8h   .. +7
        a.q[1] = ap[2];                          // K = kb+16+8h.. +7

        // load all NT B fragments for this kstep (one clause, distinct regs)
        Frag b[NT];
#pragma unroll
        for (int t = 0; t < NT; ++t) {
            const uint4* bp = bbase + ((size_t)ks * (N / 16) + t) * 64;  // 32 lanes * 2 uint4
            b[t].q[0] = bp[0];
            b[t].q[1] = bp[1];
        }
#pragma unroll
        for (int t = 0; t < NT; ++t)
            acc[t] = __builtin_amdgcn_wmma_f32_16x16x32_bf16(
                false, a.v, false, b[t].v, (short)0, acc[t], false, false);
    }

    // C/D layout: VGPR r -> row r (lanes 0-15) / row 8+r (lanes 16-31)
    float* cbase = C + (size_t)(tm * 16 + 8 * half) * N + ts * (NT * 16) + l15;
#pragma unroll
    for (int t = 0; t < NT; ++t)
#pragma unroll
        for (int r = 0; r < 8; ++r)
            cbase[r * N + t * 16] = acc[t][r];
}

// ---------------- attention dot products: a[n,h] = <h[n,h,:], att[h,:]> ------

__global__ void k_attn_dots(const float* __restrict__ h,
                            const float* __restrict__ att_s,
                            const float* __restrict__ att_d,
                            float* __restrict__ as_, float* __restrict__ ad_,
                            int N, int H, int C) {
    int n = blockIdx.x * blockDim.x + threadIdx.x;
    if (n >= N) return;
    for (int hh = 0; hh < H; ++hh) {
        const float* row = h + (size_t)n * H * C + (size_t)hh * C;
        float ss = 0.f, dd = 0.f;
        for (int c = 0; c < C; ++c) {
            float v = row[c];
            ss += v * att_s[hh * C + c];
            dd += v * att_d[hh * C + c];
        }
        as_[(size_t)n * H + hh] = ss;
        ad_[(size_t)n * H + hh] = dd;
    }
}

// ---------------- edge pass 1: segment max (ordered-uint atomicMax) ----------

__global__ void k_edge_max(const int* __restrict__ ei, int E, int N,
                           const float* __restrict__ as_, const float* __restrict__ ad_,
                           unsigned* __restrict__ mkey, int H) {
    int e = blockIdx.x * blockDim.x + threadIdx.x;
    int total = E + N;
    if (e >= total) return;
    int s, d;
    if (e < E) { s = ei[e]; d = ei[E + e]; } else { s = d = e - E; }
    for (int h = 0; h < H; ++h) {
        float v = as_[(size_t)s * H + h] + ad_[(size_t)d * H + h];
        v = v > 0.f ? v : 0.2f * v;               // leaky_relu(0.2)
        atomicMax(&mkey[(size_t)d * H + h], fkey(v));
    }
}

// ---------------- edge pass 2: exp(e - max), store weight, segment sum -------

__global__ void k_edge_expsum(const int* __restrict__ ei, int E, int N,
                              const float* __restrict__ as_, const float* __restrict__ ad_,
                              const unsigned* __restrict__ mkey,
                              float* __restrict__ ssum, float* __restrict__ wbuf, int H) {
    int e = blockIdx.x * blockDim.x + threadIdx.x;
    int total = E + N;
    if (e >= total) return;
    int s, d;
    if (e < E) { s = ei[e]; d = ei[E + e]; } else { s = d = e - E; }
    for (int h = 0; h < H; ++h) {
        float v = as_[(size_t)s * H + h] + ad_[(size_t)d * H + h];
        v = v > 0.f ? v : 0.2f * v;
        float w = __expf(v - funkey(mkey[(size_t)d * H + h]));
        wbuf[(size_t)e * H + h] = w;
        atomicAdd(&ssum[(size_t)d * H + h], w);
    }
}

// ---------------- edge pass 3: weighted gather + scatter-add -----------------
// layer 1: HC=256 channels, H=4 -> one 256-thread block per edge (coalesced)

__global__ void k_edge_agg256(const int* __restrict__ ei, int E, int N,
                              const float* __restrict__ hfeat,
                              const float* __restrict__ wbuf,
                              const float* __restrict__ ssum,
                              float* __restrict__ agg) {
    int e = blockIdx.x;
    int c = threadIdx.x;          // 0..255
    int h = c >> 6;               // c / 64
    int s, d;
    if (e < E) { s = ei[e]; d = ei[E + e]; } else { s = d = e - E; }
    float alpha = wbuf[(size_t)e * H1 + h] / (ssum[(size_t)d * H1 + h] + 1e-16f);
    atomicAdd(&agg[(size_t)d * HC1 + c], alpha * hfeat[(size_t)s * HC1 + c]);
}

// layer 2: 64 channels, H=1 -> 4 edges per 256-thread block

__global__ void k_edge_agg64(const int* __restrict__ ei, int E, int N,
                             const float* __restrict__ hfeat,
                             const float* __restrict__ wbuf,
                             const float* __restrict__ ssum,
                             float* __restrict__ agg) {
    int e = blockIdx.x * 4 + (threadIdx.x >> 6);
    int c = threadIdx.x & 63;
    int total = E + N;
    if (e >= total) return;
    int s, d;
    if (e < E) { s = ei[e]; d = ei[E + e]; } else { s = d = e - E; }
    float alpha = wbuf[e] / (ssum[d] + 1e-16f);
    atomicAdd(&agg[(size_t)d * C_DIM + c], alpha * hfeat[(size_t)s * C_DIM + c]);
}

// ---------------- epilogue: +bias, BN(eval), ELU; emit f32 and/or bf16 -------

__global__ void k_epilogue(const float* __restrict__ agg,
                           const float* __restrict__ bias,
                           const float* __restrict__ bn_g,
                           const float* __restrict__ bn_b,
                           unsigned short* __restrict__ out_bf,
                           float* __restrict__ out_f32,
                           int total, int Cdim) {
    int i = blockIdx.x * blockDim.x + threadIdx.x;
    if (i >= total) return;
    int c = i % Cdim;
    float x = agg[i] + bias[c];
    x = x * (bn_g[c] * rsqrtf(1.f + 1e-5f)) + bn_b[c];
    x = x > 0.f ? x : (__expf(x) - 1.f);          // ELU
    if (out_f32) out_f32[i] = x;
    if (out_bf)  out_bf[i]  = f32_to_bf16_rne(x);
}

// ---------------- decoder: 64 -> relu 32 -> 17 -> softmax --------------------

__global__ void k_decoder(const float* __restrict__ h,
                          const float* __restrict__ W1d, const float* __restrict__ b1,
                          const float* __restrict__ W2d, const float* __restrict__ b2,
                          float* __restrict__ out, int N) {
    int n = blockIdx.x * blockDim.x + threadIdx.x;
    if (n >= N) return;
    const float* row = h + (size_t)n * HID;
    float hid[HID / 2];
#pragma unroll
    for (int j = 0; j < HID / 2; ++j) {
        float acc = b1[j];
#pragma unroll
        for (int k = 0; k < HID; ++k) acc += row[k] * W1d[k * (HID / 2) + j];
        hid[j] = acc > 0.f ? acc : 0.f;
    }
    float lg[NCLS];
    float mx = -1e30f;
#pragma unroll
    for (int c = 0; c < NCLS; ++c) {
        float acc = b2[c];
#pragma unroll
        for (int j = 0; j < HID / 2; ++j) acc += hid[j] * W2d[j * NCLS + c];
        lg[c] = acc;
        mx = fmaxf(mx, acc);
    }
    float sum = 0.f;
#pragma unroll
    for (int c = 0; c < NCLS; ++c) { lg[c] = __expf(lg[c] - mx); sum += lg[c]; }
    float inv = 1.f / sum;
#pragma unroll
    for (int c = 0; c < NCLS; ++c) out[(size_t)n * NCLS + c] = lg[c] * inv;
}

// ---------------------------------------------------------------------------

extern "C" void kernel_launch(void* const* d_in, const int* in_sizes, int n_in,
                              void* d_out, int out_size, void* d_ws, size_t ws_size,
                              hipStream_t stream) {
    const float* x        = (const float*)d_in[0];
    const int*   ei       = (const int*)d_in[1];
    const float* W1       = (const float*)d_in[2];
    const float* att1_src = (const float*)d_in[3];
    const float* att1_dst = (const float*)d_in[4];
    const float* bias1    = (const float*)d_in[5];
    const float* bn1_g    = (const float*)d_in[6];
    const float* bn1_b    = (const float*)d_in[7];
    const float* W2       = (const float*)d_in[8];
    const float* att2_src = (const float*)d_in[9];
    const float* att2_dst = (const float*)d_in[10];
    const float* bias2    = (const float*)d_in[11];
    const float* bn2_g    = (const float*)d_in[12];
    const float* bn2_b    = (const float*)d_in[13];
    const float* dec_W1   = (const float*)d_in[14];
    const float* dec_b1   = (const float*)d_in[15];
    const float* dec_W2   = (const float*)d_in[16];
    const float* dec_b2   = (const float*)d_in[17];

    const int N = in_sizes[0] / IN_DIM;     // 50000 (divisible by 16)
    const int E = in_sizes[1] / 2;          // 400000
    const int Etot = E + N;                 // with self-loops

    // ---- workspace carve-up (256B aligned) ----
    char* p = (char*)d_ws;
    auto alloc = [&](size_t bytes) -> void* {
        void* r = (void*)p;
        p += (bytes + 255) & ~(size_t)255;
        return r;
    };
    unsigned short* xb    = (unsigned short*)alloc((size_t)N * IN_DIM * 2);
    unsigned short* W1b   = (unsigned short*)alloc((size_t)IN_DIM * HC1 * 2);
    unsigned short* W2b   = (unsigned short*)alloc((size_t)HC1 * HID * 2);
    unsigned int*   W1p   = (unsigned int*)alloc((size_t)(IN_DIM / 32) * (HC1 / 16) * 32 * 8 * 4);
    unsigned int*   W2p   = (unsigned int*)alloc((size_t)(HC1 / 32) * (HID / 16) * 32 * 8 * 4);
    float*          h1    = (float*)alloc((size_t)N * HC1 * 4);
    float*          a1s   = (float*)alloc((size_t)N * H1 * 4);
    float*          a1d   = (float*)alloc((size_t)N * H1 * 4);
    unsigned*       mkey1 = (unsigned*)alloc((size_t)N * H1 * 4);
    float*          s1    = (float*)alloc((size_t)N * H1 * 4);
    float*          wbuf1 = (float*)alloc((size_t)Etot * H1 * 4);
    float*          agg1  = (float*)alloc((size_t)N * HC1 * 4);
    unsigned short* h1ab  = (unsigned short*)alloc((size_t)N * HC1 * 2);
    float*          h2    = (float*)alloc((size_t)N * HID * 4);
    float*          a2s   = (float*)alloc((size_t)N * 4);
    float*          a2d   = (float*)alloc((size_t)N * 4);
    unsigned*       mkey2 = (unsigned*)alloc((size_t)N * 4);
    float*          s2    = (float*)alloc((size_t)N * 4);
    float*          wbuf2 = (float*)alloc((size_t)Etot * 4);
    float*          agg2  = (float*)alloc((size_t)N * HID * 4);
    float*          h2post= (float*)alloc((size_t)N * HID * 4);

    // ---- zero-init accumulators (graph-capturable async memsets) ----
    hipMemsetAsync(mkey1, 0, (size_t)N * H1 * 4, stream);   // key 0 == "below -inf"
    hipMemsetAsync(s1,    0, (size_t)N * H1 * 4, stream);
    hipMemsetAsync(agg1,  0, (size_t)N * HC1 * 4, stream);
    hipMemsetAsync(mkey2, 0, (size_t)N * 4, stream);
    hipMemsetAsync(s2,    0, (size_t)N * 4, stream);
    hipMemsetAsync(agg2,  0, (size_t)N * HID * 4, stream);

    // ---- bf16 conversions + B-fragment packing ----
    {
        int n = N * IN_DIM;
        k_f32_to_bf16<<<(n + 255) / 256, 256, 0, stream>>>(x, xb, n);
        n = IN_DIM * HC1;
        k_f32_to_bf16<<<(n + 255) / 256, 256, 0, stream>>>(W1, W1b, n);
        n = HC1 * HID;
        k_f32_to_bf16<<<(n + 255) / 256, 256, 0, stream>>>(W2, W2b, n);
        int p1 = (IN_DIM / 32) * (HC1 / 16) * 32;   // (ks,tn,lane) triples
        k_pack_b<<<(p1 + 255) / 256, 256, 0, stream>>>(W1b, W1p, IN_DIM, HC1);
        int p2 = (HC1 / 32) * (HID / 16) * 32;
        k_pack_b<<<(p2 + 255) / 256, 256, 0, stream>>>(W2b, W2p, HC1, HID);
    }

    // ---- layer 1: h1 = x @ W1 (WMMA, 16x64 strip per wave) ----
    {
        int totalWaves = (N / 16) * (HC1 / 64);            // strips of 4 N-tiles
        int blocks = (totalWaves + 7) / 8;                 // 8 waves / 256-thread block
        k_wmma_gemm<IN_DIM, HC1, 4><<<blocks, 256, 0, stream>>>(xb, W1p, h1, totalWaves);
    }
    k_attn_dots<<<(N + 255) / 256, 256, 0, stream>>>(h1, att1_src, att1_dst,
                                                     a1s, a1d, N, H1, C_DIM);
    k_edge_max<<<(Etot + 255) / 256, 256, 0, stream>>>(ei, E, N, a1s, a1d, mkey1, H1);
    k_edge_expsum<<<(Etot + 255) / 256, 256, 0, stream>>>(ei, E, N, a1s, a1d,
                                                          mkey1, s1, wbuf1, H1);
    k_edge_agg256<<<Etot, 256, 0, stream>>>(ei, E, N, h1, wbuf1, s1, agg1);
    k_epilogue<<<((size_t)N * HC1 + 255) / 256, 256, 0, stream>>>(
        agg1, bias1, bn1_g, bn1_b, h1ab, nullptr, N * HC1, HC1);

    // ---- layer 2: h2 = elu(bn(agg1)) @ W2 (WMMA, whole 64-wide row per wave) ----
    {
        int totalWaves = (N / 16) * (HID / 64);
        int blocks = (totalWaves + 7) / 8;
        k_wmma_gemm<HC1, HID, 4><<<blocks, 256, 0, stream>>>(h1ab, W2p, h2, totalWaves);
    }
    k_attn_dots<<<(N + 255) / 256, 256, 0, stream>>>(h2, att2_src, att2_dst,
                                                     a2s, a2d, N, 1, C_DIM);
    k_edge_max<<<(Etot + 255) / 256, 256, 0, stream>>>(ei, E, N, a2s, a2d, mkey2, 1);
    k_edge_expsum<<<(Etot + 255) / 256, 256, 0, stream>>>(ei, E, N, a2s, a2d,
                                                          mkey2, s2, wbuf2, 1);
    k_edge_agg64<<<(Etot + 3) / 4, 256, 0, stream>>>(ei, E, N, h2, wbuf2, s2, agg2);
    k_epilogue<<<((size_t)N * HID + 255) / 256, 256, 0, stream>>>(
        agg2, bias2, bn2_g, bn2_b, nullptr, h2post, N * HID, HID);

    // ---- decoder + softmax -> d_out [N,17] f32 ----
    k_decoder<<<(N + 127) / 128, 128, 0, stream>>>(h2post, dec_W1, dec_b1,
                                                   dec_W2, dec_b2, (float*)d_out, N);
}
